// GINEClassifier_14937896256087
// MI455X (gfx1250) — compile-verified, matrix operand order checked by reading, not run on hardware
//
#include <hip/hip_runtime.h>
#include <hip/hip_bf16.h>

#define NN_NODES 50000
#define NN_EDGES 800000
#define NN_GRAPHS 512
#define HID 128
#define EDGE_DIM 95
#define EDGE_KPAD 96
#define BN_EPS 1e-5f

#define EWAVES 4                         // waves per block in edge kernel
#define ETILE_BYTES (16 * EDGE_DIM * 4)  // 6080 B, contiguous 16-edge tile
#define ECHUNKS (ETILE_BYTES / 16)       // 380 b128 chunks
#define EITERS 12                        // ceil(380/32)

typedef __attribute__((ext_vector_type(16))) __bf16 v16bf;
typedef __attribute__((ext_vector_type(8)))  float  v8f;

// K index within a 32-wide chunk for element-pair v (0..7) of a 16-bit A/B
// fragment, per CDNA5 ISA 7.12.2 (lane half selects +8).
__device__ __forceinline__ int frag_k(int v, int half) {
    return (v < 4) ? (half * 8 + 2 * v) : (16 + half * 8 + 2 * (v - 4));
}

// A fragment (16x32 bf16) from an fp32 row (lane's own row), K-guarded.
__device__ __forceinline__ v16bf load_a_frag_f32(const float* row,
                                                 int base_k, int kmax) {
    int half = (threadIdx.x >> 4) & 1;
    v16bf a;
#pragma unroll
    for (int v = 0; v < 8; ++v) {
        int k = base_k + frag_k(v, half);
        float f0 = (k     < kmax) ? row[k]     : 0.f;
        float f1 = (k + 1 < kmax) ? row[k + 1] : 0.f;
        a[2 * v]     = (__bf16)f0;
        a[2 * v + 1] = (__bf16)f1;
    }
    return a;
}

// A fragment from LDS bf16 row (lane's own row).
__device__ __forceinline__ v16bf load_a_frag_lds(const __bf16* row, int base_k) {
    int half = (threadIdx.x >> 4) & 1;
    v16bf a;
#pragma unroll
    for (int v = 0; v < 8; ++v) {
        int k = base_k + frag_k(v, half);
        a[2 * v]     = row[k];
        a[2 * v + 1] = row[k + 1];
    }
    return a;
}

// B fragment (32x16 bf16): B[k, n] = W[n][k] with W staged in LDS as
// [out_channel][k], row stride ks. lane%16 selects the column n.
__device__ __forceinline__ v16bf load_b_frag_lds(const __bf16* __restrict__ W,
                                                 int n0, int ks, int base_k) {
    int lane = threadIdx.x & 31;
    int half = lane >> 4;
    const __bf16* row = W + (n0 + (lane & 15)) * ks;
    v16bf b;
#pragma unroll
    for (int v = 0; v < 8; ++v) {
        int k = base_k + frag_k(v, half);
        b[2 * v]     = row[k];
        b[2 * v + 1] = row[k + 1];
    }
    return b;
}

// ---------------------------------------------------------------------------
__global__ void copy_f32(const float* __restrict__ src, float* __restrict__ dst, int n) {
    int i = blockIdx.x * blockDim.x + threadIdx.x;
    if (i < n) dst[i] = src[i];
}

__global__ void zero_f32(float* __restrict__ dst, int n) {
    int i = blockIdx.x * blockDim.x + threadIdx.x;
    if (i < n) dst[i] = 0.f;
}

// ---------------------------------------------------------------------------
// Fused edge kernel: ea = edge_attr @ epw^T + epb (WMMA bf16), msg =
// relu(h[src] + ea), aggr[dst] += msg.  One wave per 16-edge tile,
// grid-stride over tiles.  The contiguous 6080-B edge_attr tile is streamed
// LDS-direct with GLOBAL_LOAD_ASYNC_TO_LDS_B128 and double-buffered on
// ASYNCcnt (12 in-order b128 issues per tile -> s_wait_asynccnt 12 releases
// the previous buffer while the next tile is in flight).
__global__ void gine_edge_msg(const float* __restrict__ edge_attr,
                              const float* __restrict__ epw,   // [128,95]
                              const float* __restrict__ epb,   // [128]
                              const int*   __restrict__ esrc,
                              const int*   __restrict__ edst,
                              const float* __restrict__ hin,   // [N,128]
                              float*       __restrict__ aggr)  // [N,128]
{
    __shared__ __bf16 Wlds[HID * EDGE_KPAD]; // epw^T as [c][k], k padded to 96
    __shared__ __align__(16) float Tile[EWAVES][2][16 * EDGE_DIM];

    for (int i = threadIdx.x; i < HID * EDGE_KPAD; i += blockDim.x) {
        int c = i / EDGE_KPAD, k = i % EDGE_KPAD;
        Wlds[i] = (k < EDGE_DIM) ? (__bf16)epw[c * EDGE_DIM + k] : (__bf16)0.f;
    }
    __syncthreads();

    const int wave = threadIdx.x >> 5;
    const int lane = threadIdx.x & 31;
    const int half = lane >> 4;
    const int n    = lane & 15;

    const int NTILES  = NN_EDGES / 16;
    const int wstride = gridDim.x * EWAVES;
    const int tile0   = blockIdx.x * EWAVES + wave;
    if (tile0 >= NTILES) return;

    // Issue the async tile copy: 12 x b128 per tile (last partially masked).
    auto prefetch = [&](int tile, int b) {
        unsigned lds0 = (unsigned)(size_t)(&Tile[wave][b][0]);
        int goff = tile * ETILE_BYTES;
#pragma unroll
        for (int it = 0; it < EITERS; ++it) {
            int c = it * 32 + lane;
            if (c < ECHUNKS) {
                unsigned ldsa = lds0 + (unsigned)c * 16u;
                int      ga   = goff + c * 16;
                asm volatile("global_load_async_to_lds_b128 %0, %1, %2"
                             :
                             : "v"(ldsa), "v"(ga), "s"(edge_attr)
                             : "memory");
            }
        }
    };

    int cur = 0;
    prefetch(tile0, 0);

    for (int tile = tile0; tile < NTILES; tile += wstride) {
        int  nxt = tile + wstride;
        bool pf  = (nxt < NTILES);
        if (pf) {
            prefetch(nxt, cur ^ 1);
            asm volatile("s_wait_asynccnt 0xc" ::: "memory"); // current buf done
        } else {
            asm volatile("s_wait_asynccnt 0x0" ::: "memory");
        }

        const float* arow = &Tile[wave][cur][n * EDGE_DIM];

        v8f acc[8];
#pragma unroll
        for (int nt = 0; nt < 8; ++nt) acc[nt] = (v8f)(0.f);

#pragma unroll
        for (int kc = 0; kc < 3; ++kc) {
            v16bf a = load_a_frag_f32(arow, kc * 32, EDGE_DIM);
#pragma unroll
            for (int nt = 0; nt < 8; ++nt) {
                v16bf b = load_b_frag_lds(Wlds, nt * 16, EDGE_KPAD, kc * 32);
                acc[nt] = __builtin_amdgcn_wmma_f32_16x16x32_bf16(
                    false, a, false, b, (short)0, acc[nt], false, false);
            }
        }

        int e0 = tile * 16;
#pragma unroll
        for (int r = 0; r < 8; ++r) {
            int e = e0 + half * 8 + r;
            size_t srow = (size_t)esrc[e] * HID;
            size_t drow = (size_t)edst[e] * HID;
#pragma unroll
            for (int nt = 0; nt < 8; ++nt) {
                int col = nt * 16 + n;
                float v = acc[nt][r] + epb[col] + hin[srow + col];
                v = fmaxf(v, 0.f);
                atomicAdd(&aggr[drow + col], v);
            }
        }
        cur ^= 1;
    }
}

// ---------------------------------------------------------------------------
// Node MLP: h = relu(BN(relu(z@w1^T+b1)@w2^T + b2)).  One wave per 16-node
// tile, w1/w2 staged bf16 in LDS (w2 overwrites w1), intermediate bounced
// through per-wave LDS staging to re-layout C -> A fragments.
__global__ void gine_node_mlp(const float* __restrict__ z,    // [N,128]
                              const float* __restrict__ w1,
                              const float* __restrict__ b1,
                              const float* __restrict__ w2,
                              const float* __restrict__ b2,
                              const float* __restrict__ bng,
                              const float* __restrict__ bnb,
                              const float* __restrict__ bnm,
                              const float* __restrict__ bnv,
                              float*       __restrict__ hout)
{
    __shared__ __bf16 Wlds[HID * HID];        // 32 KB
    __shared__ __bf16 Tst[8][16 * HID];       // 32 KB (per-wave t staging)

    int wave = threadIdx.x >> 5;
    int lane = threadIdx.x & 31;
    int tile = blockIdx.x * 8 + wave;
    int n0   = tile * 16;
    bool active = n0 < NN_NODES;
    int half = lane >> 4;
    int n    = lane & 15;

    for (int i = threadIdx.x; i < HID * HID; i += blockDim.x)
        Wlds[i] = (__bf16)w1[i];
    __syncthreads();

    if (active) {
        v8f acc[8];
#pragma unroll
        for (int nt = 0; nt < 8; ++nt) acc[nt] = (v8f)(0.f);
        const float* zrow = z + (size_t)(n0 + n) * HID;
#pragma unroll
        for (int kc = 0; kc < 4; ++kc) {
            v16bf a = load_a_frag_f32(zrow, kc * 32, HID);
#pragma unroll
            for (int nt = 0; nt < 8; ++nt) {
                v16bf b = load_b_frag_lds(Wlds, nt * 16, HID, kc * 32);
                acc[nt] = __builtin_amdgcn_wmma_f32_16x16x32_bf16(
                    false, a, false, b, (short)0, acc[nt], false, false);
            }
        }
#pragma unroll
        for (int r = 0; r < 8; ++r) {
            int M = half * 8 + r;
#pragma unroll
            for (int nt = 0; nt < 8; ++nt) {
                int col = nt * 16 + n;
                float v = fmaxf(acc[nt][r] + b1[col], 0.f);
                Tst[wave][M * HID + col] = (__bf16)v;
            }
        }
    }
    __syncthreads();

    for (int i = threadIdx.x; i < HID * HID; i += blockDim.x)
        Wlds[i] = (__bf16)w2[i];
    __syncthreads();

    if (!active) return;

    v8f acc2[8];
#pragma unroll
    for (int nt = 0; nt < 8; ++nt) acc2[nt] = (v8f)(0.f);
    const __bf16* trow = &Tst[wave][n * HID];
#pragma unroll
    for (int kc = 0; kc < 4; ++kc) {
        v16bf a = load_a_frag_lds(trow, kc * 32);
#pragma unroll
        for (int nt = 0; nt < 8; ++nt) {
            v16bf b = load_b_frag_lds(Wlds, nt * 16, HID, kc * 32);
            acc2[nt] = __builtin_amdgcn_wmma_f32_16x16x32_bf16(
                false, a, false, b, (short)0, acc2[nt], false, false);
        }
    }
#pragma unroll
    for (int r = 0; r < 8; ++r) {
        int M = half * 8 + r;
#pragma unroll
        for (int nt = 0; nt < 8; ++nt) {
            int col = nt * 16 + n;
            float v = acc2[nt][r] + b2[col];
            v = (v - bnm[col]) * (bng[col] * rsqrtf(bnv[col] + BN_EPS)) + bnb[col];
            v = fmaxf(v, 0.f);
            hout[(size_t)(n0 + M) * HID + col] = v;
        }
    }
}

// ---------------------------------------------------------------------------
// global_max_pool: values are post-ReLU (>= 0) so u32-bit atomicMax on the
// float bit pattern is order-preserving; init-0 matches the empty-graph guard.
__global__ void pool_max(const float* __restrict__ h, const int* __restrict__ batch,
                         float* __restrict__ g) {
    int i = blockIdx.x * blockDim.x + threadIdx.x;
    if (i >= NN_NODES * HID) return;
    int node = i >> 7, col = i & 127;
    unsigned int bits = __float_as_uint(h[i]);
    atomicMax((unsigned int*)&g[(size_t)batch[node] * HID + col], bits);
}

__global__ void head1(const float* __restrict__ g, const float* __restrict__ w,
                      const float* __restrict__ b, float* __restrict__ g1) {
    int i = blockIdx.x * blockDim.x + threadIdx.x;
    if (i >= NN_GRAPHS * HID) return;
    int row = i >> 7, c = i & 127;
    const float* gr = g + (size_t)row * HID;
    const float* wr = w + (size_t)c * HID;
    float acc = b[c];
#pragma unroll 8
    for (int k = 0; k < HID; ++k) acc += gr[k] * wr[k];
    g1[i] = fmaxf(acc, 0.f);
}

__global__ void head2(const float* __restrict__ g1, const float* __restrict__ w,
                      const float* __restrict__ b, float* __restrict__ out) {
    int i = blockIdx.x * blockDim.x + threadIdx.x;
    if (i >= NN_GRAPHS * 2) return;
    int row = i >> 1, o = i & 1;
    const float* gr = g1 + (size_t)row * HID;
    const float* wr = w + (size_t)o * HID;
    float acc = b[o];
#pragma unroll 8
    for (int k = 0; k < HID; ++k) acc += gr[k] * wr[k];
    out[i] = acc;
}

// ---------------------------------------------------------------------------
extern "C" void kernel_launch(void* const* d_in, const int* in_sizes, int n_in,
                              void* d_out, int out_size, void* d_ws, size_t ws_size,
                              hipStream_t stream) {
    const float* x          = (const float*)d_in[0];
    const int*   edge_index = (const int*)d_in[1];   // [2, E]
    const float* edge_attr  = (const float*)d_in[2];
    const int*   batch      = (const int*)d_in[3];
    const float* epw        = (const float*)d_in[4]; // [2,128,95]
    const float* epb        = (const float*)d_in[5]; // [2,128]
    const float* w1         = (const float*)d_in[6]; // [2,128,128]
    const float* b1         = (const float*)d_in[7];
    const float* w2         = (const float*)d_in[8];
    const float* b2         = (const float*)d_in[9];
    const float* bng        = (const float*)d_in[10];
    const float* bnb        = (const float*)d_in[11];
    const float* bnm        = (const float*)d_in[12];
    const float* bnv        = (const float*)d_in[13];
    const float* lin1_w     = (const float*)d_in[14];
    const float* lin1_b     = (const float*)d_in[15];
    const float* lin2_w     = (const float*)d_in[16];
    const float* lin2_b     = (const float*)d_in[17];
    float* out = (float*)d_out;

    const int* esrc = edge_index;
    const int* edst = edge_index + NN_EDGES;

    // workspace layout (fp32): bufA | bufB | gpool | g1
    float* bufA  = (float*)d_ws;
    float* bufB  = bufA + (size_t)NN_NODES * HID;
    float* gpool = bufB + (size_t)NN_NODES * HID;
    float* g1    = gpool + (size_t)NN_GRAPHS * HID;

    const int nelem = NN_NODES * HID;
    const int edge_blocks = 1024;                      // grid-stride over 50000 tiles
    const int mlp_blocks  = (NN_NODES / 16 + 7) / 8;   // 391

    const float* h = x;
    for (int L = 0; L < 2; ++L) {
        copy_f32<<<(nelem + 255) / 256, 256, 0, stream>>>(h, bufA, nelem);
        gine_edge_msg<<<edge_blocks, EWAVES * 32, 0, stream>>>(
            edge_attr, epw + (size_t)L * HID * EDGE_DIM, epb + (size_t)L * HID,
            esrc, edst, h, bufA);
        gine_node_mlp<<<mlp_blocks, 256, 0, stream>>>(
            bufA,
            w1 + (size_t)L * HID * HID, b1 + (size_t)L * HID,
            w2 + (size_t)L * HID * HID, b2 + (size_t)L * HID,
            bng + (size_t)L * HID, bnb + (size_t)L * HID,
            bnm + (size_t)L * HID, bnv + (size_t)L * HID,
            bufB);
        h = bufB;
    }

    zero_f32<<<(NN_GRAPHS * HID + 255) / 256, 256, 0, stream>>>(gpool, NN_GRAPHS * HID);
    pool_max<<<(nelem + 255) / 256, 256, 0, stream>>>(bufB, batch, gpool);
    head1<<<(NN_GRAPHS * HID + 255) / 256, 256, 0, stream>>>(gpool, lin1_w, lin1_b, g1);
    head2<<<(NN_GRAPHS * 2 + 255) / 256, 256, 0, stream>>>(g1, lin2_w, lin2_b, out);
}
